// MultiheadLocalAttentionV1_23888608100407
// MI455X (gfx1250) — matrix operand order repaired and checked by reading, not compiled, not run on hardware
//
#include <hip/hip_runtime.h>
#include <math.h>

// ---------------------------------------------------------------------------
// MultiheadLocalAttention for MI455X (gfx1250, wave32, WMMA 16x16x32 f16->f32)
//
// Problem: N=2, C=256, 32x32 spatial, 8 heads x 32 dim, 15x15 window (225 taps)
// Dense GEMM pieces (QKV proj, rel logits, final proj) -> v_wmma_f32_16x16x32_f16
// Attention core (banded gather + softmax + weighted sum)  -> f32 VALU + LDS
// ---------------------------------------------------------------------------

typedef __attribute__((ext_vector_type(16))) _Float16 v16h;
typedef __attribute__((ext_vector_type(8)))  float    v8f;

#define N_    2
#define C_    256
#define HW_   1024      // 32*32
#define NH_   8         // heads
#define HD_   32        // per-head dim
#define WS_   15
#define WS2_  225
#define TINV  0.17677669529663687f   // 1/sqrt(32)

#define OUT_ELEMS  (HW_ * N_ * C_)                   // 524288 floats (out)
#define ATTN_ELEMS ((size_t)N_ * NH_ * WS2_ * HW_)   // 3686400 floats (attn)

// --------- WMMA per-lane gather helpers (layouts per CDNA5 ISA 7.12.2) ------
// A 16x32 f16: lane holds row M=lane&15; K(e) = e + (e<8?0:8) + (lane<16?0:8)
// B 32x16 f16: lane holds col N=lane&15; K(e) = e + (e<8?0:8) + (lane<16?0:8)
// C/D 16x16 f32: elem i -> row i + (lane<16?0:8), col lane&15

__device__ __forceinline__ int a_koff(int lane, int e) {
    return e + ((e < 8) ? 0 : 8) + ((lane < 16) ? 0 : 8);
}
__device__ __forceinline__ int b_koff(int lane, int e) {
    return e + ((e < 8) ? 0 : 8) + ((lane < 16) ? 0 : 8);
}

// ---------------------------------------------------------------------------
// Kernel 1: fused Q/K/V 1x1-conv projections.
// One wave per 16x16 output tile of O[o,p] = sum_c W[o,c]*X[n,c,p] + b[o].
// wave id = ((mat*N + n)*16 + mtile)*64 + ntile   (3*2*16*64 = 6144 waves)
// ---------------------------------------------------------------------------
__global__ void qkv_proj_kernel(const float* __restrict__ q,
                                const float* __restrict__ k,
                                const float* __restrict__ v,
                                const float* __restrict__ WQ, const float* __restrict__ bQ,
                                const float* __restrict__ WK, const float* __restrict__ bK,
                                const float* __restrict__ WV, const float* __restrict__ bV,
                                float* __restrict__ qp, float* __restrict__ kp,
                                float* __restrict__ vp)
{
    int wave = blockIdx.x * (blockDim.x >> 5) + (threadIdx.x >> 5);
    int lane = threadIdx.x & 31;
    int nt  = wave & 63;  wave >>= 6;
    int mt  = wave & 15;  wave >>= 4;
    int n   = wave & 1;   wave >>= 1;
    int mat = wave;                           // 0=Q 1=K 2=V

    const float* X; const float* W; const float* b; float* O;
    if (mat == 0)      { X = q; W = WQ; b = bQ; O = qp; }
    else if (mat == 1) { X = k; W = WK; b = bK; O = kp; }
    else               { X = v; W = WV; b = bV; O = vp; }

    int m16 = lane & 15;
    int rowHi = (lane < 16) ? 0 : 8;
    int o = mt * 16 + m16;          // A row (output channel)
    int p = nt * 16 + m16;          // B col (spatial position)

    v8f acc = {};
#pragma unroll
    for (int kb = 0; kb < C_; kb += 32) {
        v16h a, bb;
#pragma unroll
        for (int e = 0; e < 16; ++e)
            a[e] = (_Float16)W[o * C_ + kb + a_koff(lane, e)];
#pragma unroll
        for (int e = 0; e < 16; ++e)
            bb[e] = (_Float16)X[((size_t)n * C_ + kb + b_koff(lane, e)) * HW_ + p];
        acc = __builtin_amdgcn_wmma_f32_16x16x32_f16(false, a, false, bb,
                                                     (short)0, acc, false, false);
    }
#pragma unroll
    for (int i = 0; i < 8; ++i) {
        int row = mt * 16 + i + rowHi;
        O[((size_t)n * C_ + row) * HW_ + nt * 16 + m16] = acc[i] + b[row];
    }
}

// ---------------------------------------------------------------------------
// Kernel 2: grouped rel-pos logits. Per (n,head):
//   rel[s,p] = sum_c Wrk[h,s,c] * qp[n, h*32+c, p] + brk[h,s]
// GEMM M=225(pad 240) x N=1024 x K=32 -> single WMMA per tile.
// wave id = ((((n*8)+h)*15)+stile)*64 + ptile    (2*8*15*64 = 15360 waves)
// ---------------------------------------------------------------------------
__global__ void rel_kernel(const float* __restrict__ qp,
                           const float* __restrict__ Wrk,
                           const float* __restrict__ brk,
                           float* __restrict__ rel)
{
    int wave = blockIdx.x * (blockDim.x >> 5) + (threadIdx.x >> 5);
    int lane = threadIdx.x & 31;
    int pt = wave & 63;
    int t  = wave >> 6;
    int st = t % 15;
    int nh = t / 15;
    int h  = nh & 7;
    int n  = nh >> 3;

    int m16 = lane & 15;
    int rowHi = (lane < 16) ? 0 : 8;
    int s = st * 16 + m16;
    int p = pt * 16 + m16;

    v16h a, bb;
#pragma unroll
    for (int e = 0; e < 16; ++e)
        a[e] = (s < WS2_) ? (_Float16)Wrk[((size_t)h * WS2_ + s) * HD_ + a_koff(lane, e)]
                          : (_Float16)0.f;
#pragma unroll
    for (int e = 0; e < 16; ++e)
        bb[e] = (_Float16)qp[((size_t)n * C_ + h * HD_ + b_koff(lane, e)) * HW_ + p];

    v8f acc = {};
    acc = __builtin_amdgcn_wmma_f32_16x16x32_f16(false, a, false, bb,
                                                 (short)0, acc, false, false);
#pragma unroll
    for (int i = 0; i < 8; ++i) {
        int srow = st * 16 + i + rowHi;
        if (srow < WS2_)
            rel[(((size_t)n * NH_ + h) * WS2_ + srow) * HW_ + pt * 16 + m16] =
                acc[i] + brk[h * WS2_ + srow];
    }
}

// ---------------------------------------------------------------------------
// Kernel 3: banded qk + mask + softmax over the 225 window taps.
// One thread per (n, head, position). Writes attn directly into d_out.
// ---------------------------------------------------------------------------
__global__ void qk_softmax_kernel(const float* __restrict__ qp,
                                  const float* __restrict__ kp,
                                  const float* __restrict__ rel,
                                  float* __restrict__ attn)
{
    int tid = blockIdx.x * blockDim.x + threadIdx.x;   // 16384 total
    int p = tid & (HW_ - 1);
    int h = (tid >> 10) & 7;
    int n = tid >> 13;
    if (n >= N_) return;
    int x = p >> 5, y = p & 31;

    float qs[HD_];
#pragma unroll
    for (int c = 0; c < HD_; ++c)
        qs[c] = qp[((size_t)n * C_ + h * HD_ + c) * HW_ + p] * TINV;

    const float* relb = rel + (((size_t)n * NH_ + h) * WS2_) * HW_ + p;
    const float* kpb  = kp + ((size_t)n * C_ + h * HD_) * HW_;
    float* ab = attn + (((size_t)n * NH_ + h) * WS2_) * HW_ + p;

    float mx = -3.0e38f;
    for (int s = 0; s < WS2_; ++s) {
        int dx = s / WS_ - 7, dy = s % WS_ - 7;
        int xx = x + dx, yy = y + dy;
        bool valid = (xx >= 0) & (xx < 32) & (yy >= 0) & (yy < 32);
        float dot = 0.f;
        if (valid) {
            const float* kb = kpb + xx * 32 + yy;
#pragma unroll
            for (int c = 0; c < HD_; ++c)
                dot = fmaf(qs[c], kb[(size_t)c * HW_], dot);
        }
        float val = dot + relb[(size_t)s * HW_] - (valid ? 0.f : 1e8f);
        mx = fmaxf(mx, val);
        ab[(size_t)s * HW_] = val;
    }
    float sum = 0.f;
    for (int s = 0; s < WS2_; ++s) {
        float e = __expf(ab[(size_t)s * HW_] - mx);
        sum += e;
        ab[(size_t)s * HW_] = e;
    }
    float inv = 1.f / sum;
    for (int s = 0; s < WS2_; ++s)
        ab[(size_t)s * HW_] *= inv;
}

// ---------------------------------------------------------------------------
// Kernel 4: attention-weighted value gather:
//   outh[n,h,hd,p] = sum_s attn[s,p] * (unf_v(vp)[hd,s,p] + rel_v[h,hd,s])
// Block = one (n, head, 8-position slice); attn slice staged in LDS.
// ---------------------------------------------------------------------------
__global__ void outh_kernel(const float* __restrict__ vp,
                            const float* __restrict__ attn,
                            const float* __restrict__ rel_v,
                            float* __restrict__ outh)
{
    __shared__ float lat[WS2_ * 8];
    int blk = blockIdx.x;                 // ((n*8+h)*128 + ptile)
    int pt = blk & 127;
    int h  = (blk >> 7) & 7;
    int n  = blk >> 10;
    int p0 = pt * 8;

    const float* ab = attn + (((size_t)n * NH_ + h) * WS2_) * HW_;
    for (int idx = threadIdx.x; idx < WS2_ * 8; idx += blockDim.x) {
        int s = idx >> 3, pl = idx & 7;
        lat[idx] = ab[(size_t)s * HW_ + p0 + pl];
    }
    __syncthreads();

    int hd = threadIdx.x & 31;
    int pl = threadIdx.x >> 5;
    int p = p0 + pl;
    int x = p >> 5, y = p & 31;
    const float* vb = vp + ((size_t)n * C_ + h * HD_ + hd) * HW_;
    const float* rv = rel_v + ((size_t)h * HD_ + hd) * WS2_;

    float acc = 0.f;
    for (int s = 0; s < WS2_; ++s) {
        float a = lat[s * 8 + pl];
        int dx = s / WS_ - 7, dy = s % WS_ - 7;
        int xx = x + dx, yy = y + dy;
        float vv = ((xx >= 0) & (xx < 32) & (yy >= 0) & (yy < 32)) ? vb[xx * 32 + yy] : 0.f;
        acc = fmaf(a, vv + rv[s], acc);
    }
    outh[((size_t)n * C_ + h * HD_ + hd) * HW_ + p] = acc;
}

// ---------------------------------------------------------------------------
// Kernel 5: final projection. out[(p*N+n)*C + o] = sum_c outh[n,c,p]*Wp[o,c] + bp[o]
// One wave per 16x16 tile: A[m=p,k=c] gathered transposed from outh,
// B[k=c,n=o] from Wp.  wave id = (n*64 + ptile)*16 + otile  (2048 waves)
// ---------------------------------------------------------------------------
__global__ void final_proj_kernel(const float* __restrict__ outh,
                                  const float* __restrict__ Wp,
                                  const float* __restrict__ bp,
                                  float* __restrict__ out)
{
    int wave = blockIdx.x * (blockDim.x >> 5) + (threadIdx.x >> 5);
    int lane = threadIdx.x & 31;
    int ot = wave & 15;  wave >>= 4;
    int pt = wave & 63;  wave >>= 6;
    int n  = wave;

    int m16 = lane & 15;
    int rowHi = (lane < 16) ? 0 : 8;
    int p = pt * 16 + m16;          // A row (position)
    int o = ot * 16 + m16;          // B col (output channel)

    v8f acc = {};
#pragma unroll
    for (int kb = 0; kb < C_; kb += 32) {
        v16h a, bb;
#pragma unroll
        for (int e = 0; e < 16; ++e)
            a[e] = (_Float16)outh[((size_t)n * C_ + kb + a_koff(lane, e)) * HW_ + p];
#pragma unroll
        for (int e = 0; e < 16; ++e)
            bb[e] = (_Float16)Wp[(size_t)o * C_ + kb + b_koff(lane, e)];
        acc = __builtin_amdgcn_wmma_f32_16x16x32_f16(false, a, false, bb,
                                                     (short)0, acc, false, false);
    }
#pragma unroll
    for (int i = 0; i < 8; ++i) {
        int prow = pt * 16 + i + rowHi;
        out[((size_t)prow * N_ + n) * C_ + ot * 16 + m16] = acc[i] + bp[ot * 16 + m16];
    }
}

// ---------------------------------------------------------------------------
extern "C" void kernel_launch(void* const* d_in, const int* in_sizes, int n_in,
                              void* d_out, int out_size, void* d_ws, size_t ws_size,
                              hipStream_t stream)
{
    const float* q    = (const float*)d_in[0];
    const float* k    = (const float*)d_in[1];
    const float* v    = (const float*)d_in[2];
    const float* WQ   = (const float*)d_in[3];
    const float* bQ   = (const float*)d_in[4];
    const float* WK   = (const float*)d_in[5];
    const float* bK   = (const float*)d_in[6];
    const float* WV   = (const float*)d_in[7];
    const float* bV   = (const float*)d_in[8];
    const float* Wrk  = (const float*)d_in[9];
    const float* brk  = (const float*)d_in[10];
    const float* relv = (const float*)d_in[11];
    const float* Wp   = (const float*)d_in[12];
    const float* bp   = (const float*)d_in[13];

    float* out  = (float*)d_out;                // (hw, n, c)
    float* attn = out + OUT_ELEMS;              // (n, H, WS2, hw) — 2nd output

    float* ws   = (float*)d_ws;
    float* qp   = ws;                           // 524288 f32
    float* kp   = qp + (size_t)N_ * C_ * HW_;   // 524288 f32
    float* vp   = kp + (size_t)N_ * C_ * HW_;   // 524288 f32
    float* rel  = vp + (size_t)N_ * C_ * HW_;   // 3686400 f32
    float* outh = rel + ATTN_ELEMS;             // 524288 f32

    // 6144 waves / 8 per block
    qkv_proj_kernel<<<768, 256, 0, stream>>>(q, k, v, WQ, bQ, WK, bK, WV, bV,
                                             qp, kp, vp);
    // 15360 waves / 8 per block
    rel_kernel<<<1920, 256, 0, stream>>>(qp, Wrk, brk, rel);
    // 16384 threads
    qk_softmax_kernel<<<64, 256, 0, stream>>>(qp, kp, rel, attn);
    // one block per (n, head, 8 positions)
    outh_kernel<<<2048, 256, 0, stream>>>(vp, attn, relv, outh);
    // 2048 waves / 8 per block
    final_proj_kernel<<<256, 256, 0, stream>>>(outh, Wp, bp, out);
}
